// MultiHeadAttention_7310034338108
// MI455X (gfx1250) — matrix operand order repaired and checked by reading, not compiled, not run on hardware
//
#include <hip/hip_runtime.h>
#include <hip/hip_bf16.h>
#include <cmath>

// ---------------------------------------------------------------------------
// MI455X (gfx1250) multi-head attention, bf16 WMMA path + TDM K staging.
//   D = 1024, H = 16, d_k = 64.
// ---------------------------------------------------------------------------

#define D_MODEL   1024
#define NUM_HEADS 16
#define HEAD_DIM  64

typedef __attribute__((ext_vector_type(16))) __bf16 v16bf;
typedef __attribute__((ext_vector_type(8)))  __bf16 v8bf;
typedef __attribute__((ext_vector_type(8)))  float  v8f;
typedef __attribute__((ext_vector_type(4)))  unsigned int u32x4;
typedef __attribute__((ext_vector_type(8)))  int i32x8;
typedef __attribute__((ext_vector_type(4)))  int i32x4;

static __device__ __forceinline__ v16bf cat8(v8bf lo, v8bf hi) {
    return __builtin_shufflevector(lo, hi, 0,1,2,3,4,5,6,7,8,9,10,11,12,13,14,15);
}

static __device__ __forceinline__ v8f wmma_bf16(v16bf a, v16bf b, v8f c) {
    return __builtin_amdgcn_wmma_f32_16x16x32_bf16(
        false, a, false, b, (short)0, c, false, false);
}

static __device__ __forceinline__ v8f zero8() {
    v8f z;
#pragma unroll
    for (int i = 0; i < 8; ++i) z[i] = 0.0f;
    return z;
}

// ---------------------------------------------------------------------------
// TDM: async 2D tile load (32 rows x 64 bf16, row stride 1024 elems) -> LDS.
// D# layout per CDNA5 ISA. lds_off is the byte offset
// within the workgroup's LDS allocation (dynamic LDS starts at 0 here).
// ---------------------------------------------------------------------------
static __device__ __forceinline__ void tdm_load_k_block(const __bf16* gsrc,
                                                        unsigned lds_off)
{
    unsigned long long ga = (unsigned long long)(uintptr_t)gsrc;
    u32x4 g0 = { 1u,                                    // count=1, user mode
                 lds_off,                               // lds_addr
                 (unsigned)ga,                          // global_addr[31:0]
                 (unsigned)((ga >> 32) & 0x1FFFFFFull)  // global_addr[56:32]
                   | (2u << 30) };                      // type=2 ("image")
    i32x8 g1 = { (int)0x00010000,          // data_size=1 -> 2-byte elements
                 (int)(64u << 16),         // tensor_dim0 = 64 (lo16)
                 0,                        // dim0 hi16 = 0, tensor_dim1 lo16 = 0
                 (int)(16u | (64u << 16)), // tensor_dim1 hi16 (=1<<20), tile_dim0=64
                 32,                       // tile_dim1 = 32, tile_dim2 = 0
                 1024,                     // tensor_dim0_stride = 1024 elems
                 0, 0 };
    i32x4 gz = { 0, 0, 0, 0 };
#if __clang_major__ >= 23
    i32x8 gz8 = { 0, 0, 0, 0, 0, 0, 0, 0 };
    __builtin_amdgcn_tensor_load_to_lds(g0, g1, gz, gz, gz8, 0);
#else
    __builtin_amdgcn_tensor_load_to_lds(g0, g1, gz, gz, 0);
#endif
}

// ---------------------------------------------------------------------------
// GEMM: C[M,N] = A[M,K] * W[N,K]^T + bias[N]   (K = N = 1024)
// Block: 128 threads = 4 waves; block tile 128x64; wave tile 32x64; k-step 32.
// ---------------------------------------------------------------------------
template <bool A_IS_F32, bool OUT_F32>
__global__ __launch_bounds__(128)
void gemm_xWt_bias(const void* __restrict__ Aptr,
                   const float* __restrict__ W,
                   const float* __restrict__ bias,
                   void* __restrict__ Cptr)
{
    constexpr int TM = 128, TN = 64, TK = 32;
    __shared__ __bf16 As[TM * TK];   // 8 KB
    __shared__ __bf16 Ws[TN * TK];   // 4 KB

    const int tid  = threadIdx.x;
    const int lane = tid & 31;
    const int wave = tid >> 5;                 // 0..3
    const int bm   = blockIdx.x * TM;
    const int bn   = blockIdx.y * TN;

    const int rt     = lane & 15;
    const int ka     = (lane < 16) ? 0 : 8;    // A-fragment K base
    const int kb     = (lane < 16) ? 0 : 16;   // unused; the B fragment uses ka
    const int m_half = (lane < 16) ? 0 : 8;    // C-fragment row half

    v8f acc[2][4];
#pragma unroll
    for (int ms = 0; ms < 2; ++ms)
#pragma unroll
        for (int nt = 0; nt < 4; ++nt) acc[ms][nt] = zero8();

    for (int k0 = 0; k0 < D_MODEL; k0 += TK) {
        // ---- stage A tile (128x32) into LDS as bf16 ----
        if constexpr (A_IS_F32) {
            const float* A = (const float*)Aptr;
#pragma unroll
            for (int i = 0; i < 8; ++i) {
                int linear = tid + i * 128;            // 1024 float4 chunks
                int row = linear >> 3;
                int c4  = (linear & 7) * 4;
                float4 v = *(const float4*)(A + (size_t)(bm + row) * D_MODEL + k0 + c4);
                __bf16* dst = &As[row * TK + c4];
                dst[0] = (__bf16)v.x; dst[1] = (__bf16)v.y;
                dst[2] = (__bf16)v.z; dst[3] = (__bf16)v.w;
            }
        } else {
            const __bf16* A = (const __bf16*)Aptr;
#pragma unroll
            for (int i = 0; i < 4; ++i) {
                int linear = tid + i * 128;            // 512 v8bf chunks
                int row = linear >> 2;
                int c8  = (linear & 3) * 8;
                v8bf v = *(const v8bf*)(A + (size_t)(bm + row) * D_MODEL + k0 + c8);
                *(v8bf*)&As[row * TK + c8] = v;
            }
        }
        // ---- stage W tile (64x32) as bf16 ----
#pragma unroll
        for (int i = 0; i < 4; ++i) {
            int linear = tid + i * 128;                // 512 float4 chunks
            int row = linear >> 3;
            int c4  = (linear & 7) * 4;
            float4 v = *(const float4*)(W + (size_t)(bn + row) * D_MODEL + k0 + c4);
            __bf16* dst = &Ws[row * TK + c4];
            dst[0] = (__bf16)v.x; dst[1] = (__bf16)v.y;
            dst[2] = (__bf16)v.z; dst[3] = (__bf16)v.w;
        }

        // ---- prefetch next k tile into cache (global_prefetch_b8) ----
        if (k0 + TK < D_MODEL) {
            if constexpr (A_IS_F32)
                __builtin_prefetch((const float*)Aptr + (size_t)(bm + tid) * D_MODEL + k0 + TK, 0, 3);
            else
                __builtin_prefetch((const __bf16*)Aptr + (size_t)(bm + tid) * D_MODEL + k0 + TK, 0, 3);
            if (tid < TN)
                __builtin_prefetch(W + (size_t)(bn + tid) * D_MODEL + k0 + TK, 0, 3);
        }
        __syncthreads();

        // ---- fragments + 8 WMMAs (2 M-strips x 4 N-tiles) ----
        v16bf afrag[2];
#pragma unroll
        for (int ms = 0; ms < 2; ++ms) {
            const __bf16* arow = &As[(wave * 32 + ms * 16 + rt) * TK];
            afrag[ms] = cat8(*(const v8bf*)(arow + ka),
                             *(const v8bf*)(arow + ka + 16));
        }
#pragma unroll
        for (int nt = 0; nt < 4; ++nt) {
            v16bf bfrag = cat8(*(const v8bf*)&Ws[(nt * 16 + rt) * TK + ka],
                               *(const v8bf*)&Ws[(nt * 16 + rt) * TK + ka + 16]);
#pragma unroll
            for (int ms = 0; ms < 2; ++ms)
                acc[ms][nt] = wmma_bf16(afrag[ms], bfrag, acc[ms][nt]);
        }
        __syncthreads();
    }

    // ---- epilogue: bias add, store ----
#pragma unroll
    for (int ms = 0; ms < 2; ++ms)
#pragma unroll
        for (int nt = 0; nt < 4; ++nt) {
            int n = bn + nt * 16 + rt;
            float bv = bias[n];
#pragma unroll
            for (int r = 0; r < 8; ++r) {
                int m = bm + wave * 32 + ms * 16 + m_half + r;
                float val = acc[ms][nt][r] + bv;
                if constexpr (OUT_F32)
                    ((float*)Cptr)[(size_t)m * D_MODEL + n] = val;
                else
                    ((__bf16*)Cptr)[(size_t)m * D_MODEL + n] = (__bf16)val;
            }
        }
}

// ---------------------------------------------------------------------------
// Causal flash attention over projected bf16 Q/K/V laid out [B, S, H*64].
// grid = (B*H, S/128); block = 256 (8 waves, one 16-row q-tile per wave).
// Dynamic LDS layout (so the TDM descriptor lds_addr is a known constant):
//   [0,4096)     Ks : K block [key][d]       32x64 bf16  (TDM-loaded)
//   [4096,8192)  Vt : V block transposed [d][key] 64x32 bf16
//   [8192,16384) Ps : per-wave 16x32 P tiles
// ---------------------------------------------------------------------------
__global__ __launch_bounds__(256)
void flash_attn_kernel(const __bf16* __restrict__ Qp,
                       const __bf16* __restrict__ Kp,
                       const __bf16* __restrict__ Vp,
                       __bf16* __restrict__ Op,
                       int S)
{
    extern __shared__ __attribute__((aligned(32))) char smem[];
    __bf16* Ks = (__bf16*)(smem);
    __bf16* Vt = (__bf16*)(smem + 4096);
    __bf16* Ps = (__bf16*)(smem + 8192);

    const int tid  = threadIdx.x;
    const int lane = tid & 31;
    const int wave = tid >> 5;
    const int b    = blockIdx.x >> 4;
    const int h    = blockIdx.x & 15;
    const int q0b  = blockIdx.y * 128;
    const int q0   = q0b + wave * 16;

    const int rt     = lane & 15;
    const int ka     = (lane < 16) ? 0 : 8;
    const int kb16   = (lane < 16) ? 0 : 16;
    const int m_half = (lane < 16) ? 0 : 8;

    const size_t base = ((size_t)b * S) * D_MODEL + (size_t)h * HEAD_DIM;

    // softmax scale * log2(e), pre-folded into the Q fragment
    const float SC = 0.125f * 1.44269504088896340736f;

    v16bf qfrag[2];
#pragma unroll
    for (int ds = 0; ds < 2; ++ds) {
        const __bf16* qptr = Qp + base + (size_t)(q0 + rt) * D_MODEL + ds * 32 + ka;
        v8bf lo = *(const v8bf*)qptr;
        v8bf hi = *(const v8bf*)(qptr + 16);
#pragma unroll
        for (int j = 0; j < 8; ++j) {
            lo[j] = (__bf16)((float)lo[j] * SC);
            hi[j] = (__bf16)((float)hi[j] * SC);
        }
        qfrag[ds] = cat8(lo, hi);
    }

    // ones B-fragment: a 5th WMMA column accumulates the softmax denominator
    v16bf ones;
#pragma unroll
    for (int j = 0; j < 16; ++j) ones[j] = (__bf16)1.0f;

    float mrun[8];
    v8f oacc[5];                                  // [0..3] = O, [4] = l
#pragma unroll
    for (int r = 0; r < 8; ++r) mrun[r] = -1e30f;
#pragma unroll
    for (int dt = 0; dt < 5; ++dt) oacc[dt] = zero8();

    const int kmax = q0b + 128;
    for (int kblk = 0; kblk < kmax; kblk += 32) {
        // ---- TDM: async K block (32x64 bf16) -> LDS, issued by wave 0 ----
        if (wave == 0)
            tdm_load_k_block(Kp + base + (size_t)kblk * D_MODEL, /*lds_off=*/0u);

        // ---- manual staging of V, transposed ----
        {
            int r  = tid >> 3;                    // 0..31 key rows
            int c8 = (tid & 7) * 8;               // 8 d-chunks of 8 bf16
            v8bf vv = *(const v8bf*)(Vp + base + (size_t)(kblk + r) * D_MODEL + c8);
#pragma unroll
            for (int j = 0; j < 8; ++j) Vt[(c8 + j) * 32 + r] = vv[j];
        }
        // ---- prefetch next K/V block rows ----
        if (kblk + 32 < kmax && tid < 32) {
            __builtin_prefetch(Kp + base + (size_t)(kblk + 32 + tid) * D_MODEL, 0, 3);
            __builtin_prefetch(Vp + base + (size_t)(kblk + 32 + tid) * D_MODEL, 0, 3);
        }
        __builtin_amdgcn_s_wait_tensorcnt(0);
        __syncthreads();

        if (kblk <= q0 + 15) {                    // wave still has causal work
            // ---- scores: 4 WMMAs ----
            v8f s[2];
#pragma unroll
            for (int nt = 0; nt < 2; ++nt) {
                v8f z = zero8();
#pragma unroll
                for (int ds = 0; ds < 2; ++ds) {
                    v16bf bfrag = cat8(*(const v8bf*)&Ks[(nt * 16 + rt) * 64 + ds * 32 + ka],
                                       *(const v8bf*)&Ks[(nt * 16 + rt) * 64 + ds * 32 + ka + 16]);
                    z = wmma_bf16(qfrag[ds], bfrag, z);
                }
                s[nt] = z;
            }

            // ---- causal mask only where the diagonal cuts this block ----
            if (kblk + 31 > q0) {
#pragma unroll
                for (int nt = 0; nt < 2; ++nt) {
                    int key = kblk + nt * 16 + rt;
#pragma unroll
                    for (int r = 0; r < 8; ++r)
                        if (key > q0 + m_half + r) s[nt][r] = -1e30f;
                }
            }

            // ---- online softmax: row max via lane shuffles, l via WMMA ----
            float corrv[8];
#pragma unroll
            for (int r = 0; r < 8; ++r) {
                float mx = fmaxf(s[0][r], s[1][r]);
#pragma unroll
                for (int off = 8; off >= 1; off >>= 1)
                    mx = fmaxf(mx, __shfl_xor(mx, off, 32));
                float mnew = fmaxf(mrun[r], mx);
                corrv[r] = exp2f(mrun[r] - mnew);
                mrun[r] = mnew;
                s[0][r] = exp2f(s[0][r] - mnew);
                s[1][r] = exp2f(s[1][r] - mnew);
            }
#pragma unroll
            for (int dt = 0; dt < 5; ++dt)
#pragma unroll
                for (int r = 0; r < 8; ++r) oacc[dt][r] *= corrv[r];

            // ---- C-layout -> A-layout for P via per-wave LDS tile ----
            __bf16* pw = Ps + wave * (16 * 32);
#pragma unroll
            for (int nt = 0; nt < 2; ++nt)
#pragma unroll
                for (int r = 0; r < 8; ++r)
                    pw[(m_half + r) * 32 + nt * 16 + rt] = (__bf16)s[nt][r];

            const __bf16* prow = pw + rt * 32;
            v16bf pfrag = cat8(*(const v8bf*)(prow + ka),
                               *(const v8bf*)(prow + ka + 16));

            // ---- O += P*V (4 WMMAs) ; l += P*1 (1 WMMA) ----
#pragma unroll
            for (int dt = 0; dt < 4; ++dt) {
                v16bf vfrag = cat8(*(const v8bf*)&Vt[(dt * 16 + rt) * 32 + ka],
                                   *(const v8bf*)&Vt[(dt * 16 + rt) * 32 + ka + 16]);
                oacc[dt] = wmma_bf16(pfrag, vfrag, oacc[dt]);
            }
            oacc[4] = wmma_bf16(pfrag, ones, oacc[4]);
        }
        __syncthreads();
    }

    // ---- normalize by the WMMA-accumulated denominator, store bf16 ----
#pragma unroll
    for (int r = 0; r < 8; ++r) {
        float inv = 1.0f / oacc[4][r];
        int qrow = q0 + m_half + r;
#pragma unroll
        for (int dt = 0; dt < 4; ++dt)
            Op[base + (size_t)qrow * D_MODEL + dt * 16 + rt] =
                (__bf16)(oacc[dt][r] * inv);
    }
}

// ---------------------------------------------------------------------------
extern "C" void kernel_launch(void* const* d_in, const int* in_sizes, int n_in,
                              void* d_out, int out_size, void* d_ws, size_t ws_size,
                              hipStream_t stream)
{
    // inputs: q,k,v,mask,Wq,bq,Wk,bk,Wv,bv,Wo,bo
    const float* q  = (const float*)d_in[0];
    const float* k  = (const float*)d_in[1];
    const float* v  = (const float*)d_in[2];
    const float* Wq = (const float*)d_in[4];
    const float* bq = (const float*)d_in[5];
    const float* Wk = (const float*)d_in[6];
    const float* bk = (const float*)d_in[7];
    const float* Wv = (const float*)d_in[8];
    const float* bv = (const float*)d_in[9];
    const float* Wo = (const float*)d_in[10];
    const float* bo = (const float*)d_in[11];

    const int M = in_sizes[0] / D_MODEL;                  // B*S = 4096
    const int S = (int)(sqrt((double)in_sizes[3]) + 0.5); // 2048
    const int B = M / S;

    char* ws = (char*)d_ws;
    const size_t sz = (size_t)M * D_MODEL * sizeof(__bf16);   // 8 MB each
    __bf16* Qp = (__bf16*)(ws);
    __bf16* Kp = (__bf16*)(ws + sz);
    __bf16* Vp = (__bf16*)(ws + 2 * sz);
    __bf16* At = (__bf16*)(ws + 3 * sz);

    dim3 gblk(128);
    dim3 ggrid(M / 128, D_MODEL / 64);

    gemm_xWt_bias<true, false><<<ggrid, gblk, 0, stream>>>(q, Wq, bq, Qp);
    gemm_xWt_bias<true, false><<<ggrid, gblk, 0, stream>>>(k, Wk, bk, Kp);
    gemm_xWt_bias<true, false><<<ggrid, gblk, 0, stream>>>(v, Wv, bv, Vp);

    dim3 agrid(B * NUM_HEADS, S / 128);
    flash_attn_kernel<<<agrid, dim3(256), 16384, stream>>>(Qp, Kp, Vp, At, S);

    gemm_xWt_bias<false, true><<<ggrid, gblk, 0, stream>>>(At, Wo, bo, (float*)d_out);
}
